// MAXCA_Block_9509057593876
// MI455X (gfx1250) — matrix operand order repaired and checked by reading, not compiled, not run on hardware
//
#include <hip/hip_runtime.h>
#include <hip/hip_bf16.h>
#include <math.h>
#include <stdint.h>

typedef _Float16 f16;
typedef _Float16 v16h __attribute__((ext_vector_type(16)));
typedef _Float16 h8   __attribute__((ext_vector_type(8)));
typedef float    v8f  __attribute__((ext_vector_type(8)));

#define VOX (64 * 64 * 64) /* 262144 voxels */

// ---------------------------------------------------------------------------
// helpers
// ---------------------------------------------------------------------------
__device__ __forceinline__ v16h mk_frag(h8 lo, h8 hi) {
  v16h r;
#pragma unroll
  for (int i = 0; i < 8; i++) { r[i] = lo[i]; r[i + 8] = hi[i]; }
  return r;
}

__device__ __forceinline__ v8f wmma_f16(v16h a, v16h b, v8f c) {
  // D(f32 16x16) = A(16x32 f16) * B(32x16 f16) + C
  return __builtin_amdgcn_wmma_f32_16x16x32_f16(false, a, false, b, (short)0, c,
                                                false, false);
}

// Async global->LDS staging: each active lane moves one 16B chunk per issue.
// Tracked by ASYNCcnt; caller must __syncthreads() after.
__device__ __forceinline__ void lds_async_stage(f16* ldsDst, const f16* gSrc,
                                                int bytes, int tid) {
  unsigned lbase = (unsigned)(uintptr_t)ldsDst;  // LDS aperture: low 32b = offset
  unsigned long long gbase = (unsigned long long)(uintptr_t)gSrc;
#pragma unroll 1
  for (int i = tid * 16; i < bytes; i += 256 * 16) {
    unsigned loff = lbase + (unsigned)i;
    unsigned long long ga = gbase + (unsigned)i;
    asm volatile("global_load_async_to_lds_b128 %0, %1, off" ::"v"(loff),
                 "v"(ga)
                 : "memory");
  }
  asm volatile("s_wait_asynccnt 0x0" ::: "memory");
}

// ---------------------------------------------------------------------------
// weight f32 [taps][K][N] -> f16 transposed+padded [taps][N][K+8]
// (pad breaks LDS bank-conflict pattern; pad lanes are never read)
// ---------------------------------------------------------------------------
__global__ void k_cvt_tr(const float* __restrict__ src, f16* __restrict__ dst,
                         int K, int N, int taps) {
  int idx = blockIdx.x * 256 + threadIdx.x;
  int total = taps * K * N;
  if (idx >= total) return;
  int tap = idx / (K * N);
  int rem = idx - tap * K * N;
  int k = rem / N, n = rem - k * N;
  dst[((size_t)tap * N + n) * (K + 8) + k] = (f16)src[idx];
}

// ---------------------------------------------------------------------------
// transpose [C,V] -> [V,C] (keep f32 residual) + LayerNorm -> f16
// ---------------------------------------------------------------------------
__global__ __launch_bounds__(256) void k_prep(const float* __restrict__ xin,
                                              const float* __restrict__ g,
                                              const float* __restrict__ b,
                                              float* __restrict__ x0,
                                              f16* __restrict__ xn) {
  int v = blockIdx.x * 256 + threadIdx.x;
  float r[64];
  float m = 0.f;
#pragma unroll
  for (int c = 0; c < 64; c++) { r[c] = xin[c * VOX + v]; m += r[c]; }
  m *= (1.f / 64.f);
  float var = 0.f;
#pragma unroll
  for (int c = 0; c < 64; c++) { float t = r[c] - m; var += t * t; }
  var *= (1.f / 64.f);
  float inv = rsqrtf(var + 1e-5f);
  size_t base = (size_t)v * 64;
#pragma unroll
  for (int c = 0; c < 64; c++) {
    x0[base + c] = r[c];
    xn[base + c] = (f16)((r[c] - m) * inv * g[c] + b[c]);
  }
}

// ---------------------------------------------------------------------------
// LayerNorm over 64 channels of a strided f32 buffer -> f16 [V,64]
// ---------------------------------------------------------------------------
__global__ __launch_bounds__(256) void k_ln64(const float* __restrict__ in, int ld,
                                              int off, const float* __restrict__ g,
                                              const float* __restrict__ b,
                                              f16* __restrict__ out) {
  int v = blockIdx.x * 256 + threadIdx.x;
  const float* row = in + (size_t)v * ld + off;
  float r[64];
  float m = 0.f;
#pragma unroll
  for (int c = 0; c < 64; c++) { r[c] = row[c]; m += r[c]; }
  m *= (1.f / 64.f);
  float var = 0.f;
#pragma unroll
  for (int c = 0; c < 64; c++) { float t = r[c] - m; var += t * t; }
  var *= (1.f / 64.f);
  float inv = rsqrtf(var + 1e-5f);
  size_t o = (size_t)v * 64;
#pragma unroll
  for (int c = 0; c < 64; c++) out[o + c] = (f16)((r[c] - m) * inv * g[c] + b[c]);
}

// ---------------------------------------------------------------------------
// Generic WMMA GEMM:  out = act(A[MxK]@B + bias) (+res)
// A: f16 row-major (lda).  Bt: f16 transposed+padded [N][K+8] (async -> LDS).
// act: 0 none, 1 exact GELU.
// ---------------------------------------------------------------------------
template <int K, int NT>
__global__ __launch_bounds__(256) void k_gemm_wmma(
    const f16* __restrict__ A, int lda, const f16* __restrict__ Bt,
    const float* __restrict__ bias, const float* __restrict__ res, int ldr,
    int roff, float* __restrict__ out32, int ldo, int o32off,
    f16* __restrict__ out16, int ldo16, int o16off, int act) {
  constexpr int N = NT * 16;
  constexpr int KP = K + 8;
  __shared__ __align__(32) f16 sBt[N * KP];
  int tid = threadIdx.x;
  lds_async_stage(sBt, Bt, N * KP * 2, tid);
  __syncthreads();

  int wave = tid >> 5, lane = tid & 31, lg = lane >> 4, ln = lane & 15;
  int m0 = blockIdx.x * 128 + wave * 16;
  const f16* arow = A + (size_t)(m0 + ln) * lda;

  v8f acc[NT];
#pragma unroll
  for (int t = 0; t < NT; t++)
#pragma unroll
    for (int j = 0; j < 8; j++) acc[t][j] = 0.f;

#pragma unroll 1
  for (int kk = 0; kk < K; kk += 32) {
    h8 lo = *(const h8*)(arow + kk + lg * 8);
    h8 hi = *(const h8*)(arow + kk + lg * 8 + 16);
    v16h a = mk_frag(lo, hi);
#pragma unroll
    for (int t = 0; t < NT; t++) {
      const f16* brow = sBt + (t * 16 + ln) * KP + kk + lg * 8;
      v16h bf = mk_frag(*(const h8*)brow, *(const h8*)(brow + 16));
      acc[t] = wmma_f16(a, bf, acc[t]);
    }
  }

#pragma unroll
  for (int t = 0; t < NT; t++) {
    int n = t * 16 + ln;
    float bv = bias ? bias[n] : 0.f;
#pragma unroll
    for (int j = 0; j < 8; j++) {
      int m = m0 + j + lg * 8;
      float x = acc[t][j] + bv;
      if (act == 1) x = 0.5f * x * (1.0f + erff(x * 0.70710678118654752f));
      if (res) x += res[(size_t)m * ldr + roff + n];
      if (out32) out32[(size_t)m * ldo + o32off + n] = x;
      if (out16) out16[(size_t)m * ldo16 + o16off + n] = (f16)x;
    }
  }
}

// ---------------------------------------------------------------------------
// 3x3x3 conv (SAME) as 27-tap shifted implicit GEMM with WMMA.
// X: f16 [VOX x 64], W: f16 [27][N][72] (transposed+padded).
// act: 0 none, 2 LeakyReLU(0.2)
// ---------------------------------------------------------------------------
template <int NT>
__global__ __launch_bounds__(256) void k_conv3d_wmma(
    const f16* __restrict__ X, const f16* __restrict__ W,
    const float* __restrict__ bias, float* __restrict__ out32,
    f16* __restrict__ out16, int act) {
  constexpr int N = NT * 16;
  constexpr int KP = 72; /* 64 + 8 pad */
  __shared__ __align__(32) f16 sW[N * KP];
  int tid = threadIdx.x, wave = tid >> 5, lane = tid & 31;
  int lg = lane >> 4, ln = lane & 15;
  int m0 = blockIdx.x * 128 + wave * 16;
  int row = m0 + ln;
  int d = row & 63, w = (row >> 6) & 63, h = row >> 12;

  v8f acc[NT];
#pragma unroll
  for (int t = 0; t < NT; t++)
#pragma unroll
    for (int j = 0; j < 8; j++) acc[t][j] = 0.f;

#pragma unroll 1
  for (int tap = 0; tap < 27; tap++) {
    __syncthreads();  // previous tap's reads done before restage
    lds_async_stage(sW, W + (size_t)tap * N * KP, N * KP * 2, tid);
    __syncthreads();
    if (tap + 1 < 27) __builtin_prefetch(W + (size_t)(tap + 1) * N * KP, 0, 1);

    int nh = h + tap / 9 - 1;
    int nw = w + (tap / 3) % 3 - 1;
    int nd = d + tap % 3 - 1;
    bool valid = ((unsigned)nh < 64u) && ((unsigned)nw < 64u) && ((unsigned)nd < 64u);
    const f16* arow = X + (size_t)(((nh * 64) + nw) * 64 + nd) * 64;

#pragma unroll
    for (int kk = 0; kk < 64; kk += 32) {
      v16h a;
      if (valid) {
        h8 lo = *(const h8*)(arow + kk + lg * 8);
        h8 hi = *(const h8*)(arow + kk + lg * 8 + 16);
        a = mk_frag(lo, hi);
      } else {
#pragma unroll
        for (int e = 0; e < 16; e++) a[e] = (f16)0.f;
      }
#pragma unroll
      for (int t = 0; t < NT; t++) {
        const f16* brow = sW + (t * 16 + ln) * KP + kk + lg * 8;
        v16h bf = mk_frag(*(const h8*)brow, *(const h8*)(brow + 16));
        acc[t] = wmma_f16(a, bf, acc[t]);
      }
    }
  }

#pragma unroll
  for (int t = 0; t < NT; t++) {
    int n = t * 16 + ln;
    float bv = bias ? bias[n] : 0.f;
#pragma unroll
    for (int j = 0; j < 8; j++) {
      int m = m0 + j + lg * 8;
      float x = acc[t][j] + bv;
      if (act == 2) x = (x >= 0.f) ? x : 0.2f * x;
      if (out32) out32[(size_t)m * N + n] = x;
      if (out16) out16[(size_t)m * N + n] = (f16)x;
    }
  }
}

// ---------------------------------------------------------------------------
// Channel cross-covariance attention for one (x-position, head).
// QKV f32 [VOX x 192], temp[4].  mode 0: global (x=f, n=g); 1: local (x=g, n=f)
// Writes f16 XO [VOX x 64] (input of the output projection GEMM).
// ---------------------------------------------------------------------------
#define NP 520 /* 512 + 8 pad halves: row stride 1040B -> 4-bank row skew */

__device__ __forceinline__ size_t xca_voxel(int x, int n, int mode) {
  int g = mode ? x : n;
  int f = mode ? n : x;
  int h = ((g >> 6) & 7) * 8 + ((f >> 6) & 7);
  int w = ((g >> 3) & 7) * 8 + ((f >> 3) & 7);
  int dd = (g & 7) * 8 + (f & 7);
  return (size_t)((h * 64 + w) * 64 + dd);
}

__global__ __launch_bounds__(256) void k_xca_attn(const float* __restrict__ QKV,
                                                  const float* __restrict__ temp,
                                                  f16* __restrict__ XO, int mode) {
  __shared__ __align__(32) f16 hQ[16][NP];
  __shared__ __align__(32) f16 hK[16][NP];
  __shared__ __align__(32) f16 hV[16][NP];
  __shared__ float red[32][8];
  __shared__ float invn[32];
  __shared__ float sA[16][16];
  __shared__ __align__(32) f16 hA[16][32];

  int tid = threadIdx.x;
  int x = blockIdx.x >> 2, head = blockIdx.x & 3;

#pragma unroll 1
  for (int idx = tid; idx < 16 * 512; idx += 256) {
    int c = idx >> 9, n = idx & 511;
    const float* p = QKV + xca_voxel(x, n, mode) * 192 + head * 16 + c;
    hQ[c][n] = (f16)p[0];
    hK[c][n] = (f16)p[64];
    hV[c][n] = (f16)p[128];
  }
  __syncthreads();

  {  // row L2 norms of q (units 0..15) and k (units 16..31)
    int unit = tid >> 3, sub = tid & 7;
    int r = unit & 15, isK = unit >> 4;
    float s = 0.f;
    for (int n = sub; n < 512; n += 8) {
      float q = isK ? (float)hK[r][n] : (float)hQ[r][n];
      s += q * q;
    }
    red[unit][sub] = s;
  }
  __syncthreads();
  if (tid < 32) {
    float s = 0.f;
#pragma unroll
    for (int i = 0; i < 8; i++) s += red[tid][i];
    invn[tid] = 1.f / fmaxf(sqrtf(s), 1e-12f);
  }
  __syncthreads();
#pragma unroll 1
  for (int idx = tid; idx < 16 * 512; idx += 256) {
    int c = idx >> 9, n = idx & 511;
    hQ[c][n] = (f16)((float)hQ[c][n] * invn[c]);
    hK[c][n] = (f16)((float)hK[c][n] * invn[16 + c]);
  }
  __syncthreads();

  int wave = tid >> 5, lane = tid & 31, lg = lane >> 4, ln = lane & 15;

  if (wave == 0) {  // attn = qn @ kn^T, K = 512
    v8f acc;
#pragma unroll
    for (int j = 0; j < 8; j++) acc[j] = 0.f;
#pragma unroll 1
    for (int kk = 0; kk < 512; kk += 32) {
      h8 lo = *(const h8*)(&hQ[ln][kk + lg * 8]);
      h8 hi = *(const h8*)(&hQ[ln][kk + lg * 8 + 16]);
      v16h a = mk_frag(lo, hi);
      const f16* brow = &hK[ln][kk + lg * 8];
      v16h b = mk_frag(*(const h8*)brow, *(const h8*)(brow + 16));
      acc = wmma_f16(a, b, acc);
    }
    float ts = temp[head];
#pragma unroll
    for (int j = 0; j < 8; j++) sA[j + lg * 8][ln] = acc[j] * ts;
  }
  __syncthreads();

  if (tid < 16) {  // softmax over last dim, write f16 attn padded to K=32
    float mx = -1e30f;
#pragma unroll
    for (int c = 0; c < 16; c++) mx = fmaxf(mx, sA[tid][c]);
    float e[16], s = 0.f;
#pragma unroll
    for (int c = 0; c < 16; c++) { e[c] = expf(sA[tid][c] - mx); s += e[c]; }
    float rs = 1.f / s;
#pragma unroll
    for (int c = 0; c < 16; c++) hA[tid][c] = (f16)(e[c] * rs);
#pragma unroll
    for (int c = 16; c < 32; c++) hA[tid][c] = (f16)0.f;
  }
  __syncthreads();

  {  // out = attn @ v ; each of 8 waves does 4 n-tiles of 16
    h8 lo = *(const h8*)(&hA[ln][lg * 16]);
    h8 hi = *(const h8*)(&hA[ln][lg * 16 + 8]);
    v16h a = mk_frag(lo, hi);
#pragma unroll
    for (int t = 0; t < 4; t++) {
      int n0 = (wave * 4 + t) * 16;
      v16h b;
#pragma unroll
      for (int e = 0; e < 16; e++)
        b[e] = (lg == 0) ? hV[e][n0 + ln] : (f16)0.f;  // K rows 16..31 padded 0
      v8f acc;
#pragma unroll
      for (int j = 0; j < 8; j++) acc[j] = 0.f;
      acc = wmma_f16(a, b, acc);
      int n = n0 + ln;
      size_t vox = xca_voxel(x, n, mode);
#pragma unroll
      for (int j = 0; j < 8; j++)
        XO[vox * 64 + head * 16 + (j + lg * 8)] = (f16)acc[j];
    }
  }
}

// ---------------------------------------------------------------------------
// SE: per-channel mean, tiny MLP, sigmoid; final fuse + transpose + x_in
// ---------------------------------------------------------------------------
__global__ void k_zero64(float* p) { p[threadIdx.x] = 0.f; }

#define CS_ROWS 1024
__global__ __launch_bounds__(256) void k_colsum(const float* __restrict__ Y,
                                                float* __restrict__ acc) {
  __shared__ float s[4][64];
  int tid = threadIdx.x;
  int c = tid & 63, grp = tid >> 6;
  float p = 0.f;
  int r0 = blockIdx.x * CS_ROWS;
  for (int r = grp; r < CS_ROWS; r += 4) p += Y[(size_t)(r0 + r) * 64 + c];
  s[grp][c] = p;
  __syncthreads();
  if (tid < 64) atomicAdd(acc + tid, s[0][tid] + s[1][tid] + s[2][tid] + s[3][tid]);
}

__global__ void k_se(const float* __restrict__ msum, const float* __restrict__ w1,
                     const float* __restrict__ b1, const float* __restrict__ w2,
                     const float* __restrict__ b2, float* __restrict__ wse) {
  __shared__ float m[64], hd[16];
  int tid = threadIdx.x;  // 64 threads
  m[tid] = msum[tid] * (1.f / (float)VOX);
  __syncthreads();
  if (tid < 16) {
    float s = b1[tid];
    for (int c = 0; c < 64; c++) s += m[c] * w1[c * 16 + tid];
    hd[tid] = fmaxf(s, 0.f);
  }
  __syncthreads();
  float s = b2[tid];
  for (int r = 0; r < 16; r++) s += hd[r] * w2[r * 64 + tid];
  wse[tid] = 1.f / (1.f + expf(-s));
}

__global__ __launch_bounds__(256) void k_final(const float* __restrict__ y2,
                                               const float* __restrict__ wse,
                                               const float* __restrict__ x1,
                                               const float* __restrict__ xin,
                                               float* __restrict__ out) {
  int idx = blockIdx.x * 256 + threadIdx.x;  // over 64 * VOX, idx = c*VOX + v
  int c = idx >> 18;
  int v = idx & (VOX - 1);
  out[idx] = y2[(size_t)v * 64 + c] * wse[c] + x1[(size_t)v * 64 + c] + xin[idx];
}

// ---------------------------------------------------------------------------
// Host launcher
// ---------------------------------------------------------------------------
extern "C" void kernel_launch(void* const* d_in, const int* in_sizes, int n_in,
                              void* d_out, int out_size, void* d_ws, size_t ws_size,
                              hipStream_t stream) {
  (void)in_sizes; (void)n_in; (void)out_size; (void)ws_size;

  const float* x_in    = (const float*)d_in[0];
  const float* ln1_s   = (const float*)d_in[1];
  const float* ln1_b   = (const float*)d_in[2];
  const float* in_w    = (const float*)d_in[3];
  const float* in_b    = (const float*)d_in[4];
  const float* g_ln_s  = (const float*)d_in[5];
  const float* g_ln_b  = (const float*)d_in[6];
  const float* g_qkv_w = (const float*)d_in[7];
  const float* g_qkv_b = (const float*)d_in[8];
  const float* g_temp  = (const float*)d_in[9];
  const float* g_out_w = (const float*)d_in[10];
  const float* g_out_b = (const float*)d_in[11];
  const float* l_ln_s  = (const float*)d_in[12];
  const float* l_ln_b  = (const float*)d_in[13];
  const float* l_qkv_w = (const float*)d_in[14];
  const float* l_qkv_b = (const float*)d_in[15];
  const float* l_temp  = (const float*)d_in[16];
  const float* l_out_w = (const float*)d_in[17];
  const float* l_out_b = (const float*)d_in[18];
  const float* mx_w    = (const float*)d_in[19];
  const float* mx_b    = (const float*)d_in[20];
  const float* r_ln_s  = (const float*)d_in[21];
  const float* r_ln_b  = (const float*)d_in[22];
  const float* r_c1_w  = (const float*)d_in[23];
  const float* r_c1_b  = (const float*)d_in[24];
  const float* r_c2_w  = (const float*)d_in[25];
  const float* r_c2_b  = (const float*)d_in[26];
  const float* se1_w   = (const float*)d_in[27];
  const float* se1_b   = (const float*)d_in[28];
  const float* se2_w   = (const float*)d_in[29];
  const float* se2_b   = (const float*)d_in[30];

  char* ws = (char*)d_ws;
  size_t off = 0;
  auto carve = [&](size_t bytes) -> char* {
    char* p = ws + off;
    off += (bytes + 255) & ~(size_t)255;
    return p;
  };

  float* x0    = (float*)carve((size_t)VOX * 64 * 4);
  f16*   xn16  = (f16*)  carve((size_t)VOX * 64 * 2);
  float* y128  = (float*)carve((size_t)VOX * 128 * 4);
  f16*   n16   = (f16*)  carve((size_t)VOX * 64 * 2);
  float* qkv   = (float*)carve((size_t)VOX * 192 * 4);
  f16*   xo16  = (f16*)  carve((size_t)VOX * 64 * 2);
  f16*   uv16  = (f16*)  carve((size_t)VOX * 128 * 2);
  float* x1    = (float*)carve((size_t)VOX * 64 * 4);
  f16*   y1_16 = (f16*)  carve((size_t)VOX * 64 * 2);
  float* y2    = (float*)carve((size_t)VOX * 64 * 4);
  // transposed + padded f16 weights: [taps][N][K+8]
  f16* in_w16 = (f16*)carve((size_t)128 * 72 * 2);        // K=64,N=128
  f16* gqkv16 = (f16*)carve((size_t)27 * 192 * 72 * 2);   // K=64,N=192
  f16* gout16 = (f16*)carve((size_t)64 * 72 * 2);         // K=64,N=64
  f16* lqkv16 = (f16*)carve((size_t)27 * 192 * 72 * 2);
  f16* lout16 = (f16*)carve((size_t)64 * 72 * 2);
  f16* mx16   = (f16*)carve((size_t)64 * 136 * 2);        // K=128,N=64
  f16* rc1w16 = (f16*)carve((size_t)27 * 64 * 72 * 2);
  f16* rc2w16 = (f16*)carve((size_t)27 * 64 * 72 * 2);
  float* msum = (float*)carve(64 * 4);
  float* wse  = (float*)carve(64 * 4);

  auto cvt = [&](const float* s, f16* d, int K, int N, int taps) {
    int n = taps * K * N;
    k_cvt_tr<<<(n + 255) / 256, 256, 0, stream>>>(s, d, K, N, taps);
  };
  cvt(in_w, in_w16, 64, 128, 1);
  cvt(g_qkv_w, gqkv16, 64, 192, 27);
  cvt(g_out_w, gout16, 64, 64, 1);
  cvt(l_qkv_w, lqkv16, 64, 192, 27);
  cvt(l_out_w, lout16, 64, 64, 1);
  cvt(mx_w, mx16, 128, 64, 1);
  cvt(r_c1_w, rc1w16, 64, 64, 27);
  cvt(r_c2_w, rc2w16, 64, 64, 27);

  // ---- MultiAxisXCA ----
  k_prep<<<VOX / 256, 256, 0, stream>>>(x_in, ln1_s, ln1_b, x0, xn16);
  // y = gelu(ln(x) @ in_w + b) -> y128 [V,128] (u = cols 0..63, v = 64..127)
  k_gemm_wmma<64, 8><<<VOX / 128, 256, 0, stream>>>(
      xn16, 64, in_w16, in_b, nullptr, 0, 0, y128, 128, 0, nullptr, 0, 0, 1);

  // global branch (u)
  k_ln64<<<VOX / 256, 256, 0, stream>>>(y128, 128, 0, g_ln_s, g_ln_b, n16);
  k_conv3d_wmma<12><<<VOX / 128, 256, 0, stream>>>(n16, gqkv16, g_qkv_b, qkv,
                                                   nullptr, 0);
  k_xca_attn<<<2048, 256, 0, stream>>>(qkv, g_temp, xo16, 0);
  k_gemm_wmma<64, 4><<<VOX / 128, 256, 0, stream>>>(
      xo16, 64, gout16, g_out_b, y128, 128, 0, nullptr, 0, 0, uv16, 128, 0, 0);

  // local branch (v)
  k_ln64<<<VOX / 256, 256, 0, stream>>>(y128, 128, 64, l_ln_s, l_ln_b, n16);
  k_conv3d_wmma<12><<<VOX / 128, 256, 0, stream>>>(n16, lqkv16, l_qkv_b, qkv,
                                                   nullptr, 0);
  k_xca_attn<<<2048, 256, 0, stream>>>(qkv, l_temp, xo16, 1);
  k_gemm_wmma<64, 4><<<VOX / 128, 256, 0, stream>>>(
      xo16, 64, lout16, l_out_b, y128, 128, 64, nullptr, 0, 0, uv16, 128, 64, 0);

  // x1 = concat(u,v) @ mx_out + b + sc0
  k_gemm_wmma<128, 4><<<VOX / 128, 256, 0, stream>>>(
      uv16, 128, mx16, mx_b, x0, 64, 0, x1, 64, 0, nullptr, 0, 0, 0);

  // ---- RCAB ----
  k_ln64<<<VOX / 256, 256, 0, stream>>>(x1, 64, 0, r_ln_s, r_ln_b, n16);
  k_conv3d_wmma<4><<<VOX / 128, 256, 0, stream>>>(n16, rc1w16, r_c1_b, nullptr,
                                                  y1_16, 2);
  k_conv3d_wmma<4><<<VOX / 128, 256, 0, stream>>>(y1_16, rc2w16, r_c2_b, y2,
                                                  nullptr, 0);
  k_zero64<<<1, 64, 0, stream>>>(msum);
  k_colsum<<<VOX / CS_ROWS, 256, 0, stream>>>(y2, msum);
  k_se<<<1, 64, 0, stream>>>(msum, se1_w, se1_b, se2_w, se2_b, wse);
  k_final<<<(VOX * 64) / 256, 256, 0, stream>>>(y2, wse, x1, x_in, (float*)d_out);
}
